// GNNClassifier_47218870452965
// MI455X (gfx1250) — matrix operand order repaired and checked by reading, not compiled, not run on hardware
//
#include <hip/hip_runtime.h>
#include <hip/hip_bf16.h>

// ---------------------------------------------------------------------------
// GNN classifier: GCNConv -> 32x GatedGraphConv(GRU) -> FC, for MI455X gfx1250
// N=50000 nodes, E=1.6M edges, F_IN=1152, H=32, C=7, L=32
// ---------------------------------------------------------------------------

#define H32 32

typedef __attribute__((ext_vector_type(16))) __bf16 v16bf;
typedef __attribute__((ext_vector_type(8)))  float  v8f;
typedef __attribute__((ext_vector_type(2)))  float  v2f;

union ABf16 { v16bf v; unsigned int u[8]; };

// ---- gfx1250 async global->LDS staging (ASYNCcnt path), with fallback -----
#if defined(__AMDGCN__) && __has_builtin(__builtin_amdgcn_global_load_async_to_lds_b128)
#define HAVE_ASYNC_LDS 1
#else
#define HAVE_ASYNC_LDS 0
#endif

#if HAVE_ASYNC_LDS
typedef int vec4i __attribute__((vector_size(4 * sizeof(int))));
typedef __attribute__((address_space(1))) vec4i* g_vec4i_p;   // global pointee
typedef __attribute__((address_space(3))) vec4i* l_vec4i_p;   // LDS pointee
__device__ __forceinline__ void async_b128(const void* gsrc, void* ldst) {
  __builtin_amdgcn_global_load_async_to_lds_b128(
      (g_vec4i_p)(uintptr_t)gsrc,
      (l_vec4i_p)(unsigned int)(uintptr_t)ldst, 0, 0);
}
#if __has_builtin(__builtin_amdgcn_s_wait_asynccnt)
#define WAIT_ASYNC() __builtin_amdgcn_s_wait_asynccnt(0)
#else
#define WAIT_ASYNC() asm volatile("s_wait_asynccnt 0" ::: "memory")
#endif
#endif

__device__ __forceinline__ unsigned int f2bf_bits(float f) {
  unsigned int u = __builtin_bit_cast(unsigned int, f);
  return (u + 0x7FFFu + ((u >> 16) & 1u)) >> 16;   // RNE truncate to bf16
}
__device__ __forceinline__ unsigned int pk_bf(float lo, float hi) {
  return (f2bf_bits(lo) & 0xFFFFu) | (f2bf_bits(hi) << 16);
}

// --- prep: W1 [K][32] fp32  ->  Wt [32][K] bf16 (transposed, contiguous-K) --
__global__ void k_prep_wt(const float* __restrict__ W1,
                          unsigned short* __restrict__ Wt, int K) {
  int idx = blockIdx.x * 256 + threadIdx.x;
  if (idx < H32 * K) {
    int n = idx / K, k = idx - n * K;
    Wt[idx] = (unsigned short)f2bf_bits(W1[(size_t)k * H32 + n]);
  }
}

// --- degree / dinv ---------------------------------------------------------
__global__ void k_zero_f(float* __restrict__ p, int n) {
  int i = blockIdx.x * 256 + threadIdx.x;
  if (i < n) p[i] = 0.0f;
}
__global__ void k_count(float* __restrict__ deg, const int* __restrict__ col, int E) {
  int e = blockIdx.x * 256 + threadIdx.x;
  if (e < E) atomicAdd(&deg[col[e]], 1.0f);
}
__global__ void k_dinv(float* __restrict__ d, int n) {
  int i = blockIdx.x * 256 + threadIdx.x;
  if (i < n) d[i] = rsqrtf(d[i] + 1.0f);   // +1 self loop; deg>=1 always
}

// --- big GEMM: hlin = x @ W1, bf16 WMMA, one 16x32 tile per wave -----------
__global__ void __launch_bounds__(256) k_gemm1(const float* __restrict__ x,
    const unsigned short* __restrict__ Wt, float* __restrict__ out,
    int Nn, int K) {
  const int lane = threadIdx.x & 31;
  const int wave = threadIdx.x >> 5;
  const int tile = blockIdx.x * 8 + wave;
  const int m0   = tile * 16;
  if (m0 >= Nn) return;                       // wave-uniform exit, EXEC stays full
  const int half = lane >> 4;                 // 0: lanes 0-15, 1: lanes 16-31
  const int ln   = lane & 15;
  int rowA = m0 + ln; if (rowA > Nn - 1) rowA = Nn - 1;
  const float* __restrict__ xr = x + (size_t)rowA * K;
  const unsigned short* __restrict__ bp0 = Wt + (size_t)ln * K + half * 16;
  const unsigned short* __restrict__ bp1 = Wt + (size_t)(16 + ln) * K + half * 16;

  v8f acc0 = {}; v8f acc1 = {};
  for (int k0 = 0; k0 < K; k0 += 32) {
    // A tile 16x32 bf16: lane<16 -> K {k0..k0+15}; lane>=16 -> K {k0+16..k0+31}
    const float4* pa = (const float4*)(xr + k0 + half * 16);
    float4 f0 = pa[0], f1 = pa[1];
    const float4* pb = (const float4*)(xr + k0 + half * 16 + 8);
    float4 f2 = pb[0], f3 = pb[1];
    __builtin_prefetch(xr + k0 + 64, 0, 1);   // global_prefetch_b8 for next chunk
    ABf16 a;
    a.u[0] = pk_bf(f0.x, f0.y); a.u[1] = pk_bf(f0.z, f0.w);
    a.u[2] = pk_bf(f1.x, f1.y); a.u[3] = pk_bf(f1.z, f1.w);
    a.u[4] = pk_bf(f2.x, f2.y); a.u[5] = pk_bf(f2.z, f2.w);
    a.u[6] = pk_bf(f3.x, f3.y); a.u[7] = pk_bf(f3.z, f3.w);
    // B tile 32x16 bf16: per-lane 16 consecutive K (contiguous in transposed Wt)
    ABf16 b0, b1;
    const uint4* q0 = (const uint4*)(bp0 + k0);
    uint4 t0 = q0[0], t1 = q0[1];
    b0.u[0]=t0.x; b0.u[1]=t0.y; b0.u[2]=t0.z; b0.u[3]=t0.w;
    b0.u[4]=t1.x; b0.u[5]=t1.y; b0.u[6]=t1.z; b0.u[7]=t1.w;
    const uint4* q1 = (const uint4*)(bp1 + k0);
    uint4 t2 = q1[0], t3 = q1[1];
    b1.u[0]=t2.x; b1.u[1]=t2.y; b1.u[2]=t2.z; b1.u[3]=t2.w;
    b1.u[4]=t3.x; b1.u[5]=t3.y; b1.u[6]=t3.z; b1.u[7]=t3.w;

    acc0 = __builtin_amdgcn_wmma_f32_16x16x32_bf16(false, a.v, false, b0.v,
                                                   (short)0, acc0, false, false);
    acc1 = __builtin_amdgcn_wmma_f32_16x16x32_bf16(false, a.v, false, b1.v,
                                                   (short)0, acc1, false, false);
  }
  // D: vgpr r -> row m0 + r + 8*half, col ln / 16+ln
  float* __restrict__ dst = out + (size_t)(m0 + half * 8) * H32 + ln;
  if (m0 + 16 <= Nn) {                        // full tile: branchless stores
#pragma unroll
    for (int r = 0; r < 8; ++r) {
      dst[(size_t)r * H32]      = acc0[r];
      dst[(size_t)r * H32 + 16] = acc1[r];
    }
  } else {
#pragma unroll
    for (int r = 0; r < 8; ++r) {
      int mm = m0 + half * 8 + r;
      if (mm < Nn) {
        out[(size_t)mm * H32 + ln]      = acc0[r];
        out[(size_t)mm * H32 + 16 + ln] = acc1[r];
      }
    }
  }
}

// --- GCN: self-loop init, edge scatter with deg^{-1/2} norm, bias+relu -----
__global__ void k_gcn_self(const float* __restrict__ hlin,
                           const float* __restrict__ dinv,
                           float* __restrict__ hg, int Nn) {
  int idx = blockIdx.x * 256 + threadIdx.x;
  if (idx < Nn * H32) {
    int i = idx >> 5;
    float d = dinv[i];
    hg[idx] = d * d * hlin[idx];
  }
}
__global__ void k_gcn_scatter(const float* __restrict__ hlin,
                              float* __restrict__ hg,
                              const float* __restrict__ dinv,
                              const int* __restrict__ row,
                              const int* __restrict__ col, int E) {
  int idx = blockIdx.x * 256 + threadIdx.x;   // E*8 threads, float4 per thread
  int e = idx >> 3;
  if (e >= E) return;
  int q = (idx & 7) * 4;
  int r = row[e], c = col[e];
  float nrm = dinv[r] * dinv[c];
  float4 v = *(const float4*)(hlin + (size_t)r * H32 + q);
  float* dst = hg + (size_t)c * H32 + q;
  atomicAdd(dst + 0, v.x * nrm);
  atomicAdd(dst + 1, v.y * nrm);
  atomicAdd(dst + 2, v.z * nrm);
  atomicAdd(dst + 3, v.w * nrm);
}
__global__ void k_bias_relu(const float* __restrict__ hg,
                            const float* __restrict__ b1,
                            float* __restrict__ h, int n32) {
  int idx = blockIdx.x * 256 + threadIdx.x;
  if (idx < n32) h[idx] = fmaxf(hg[idx] + b1[idx & 31], 0.0f);
}

// --- per-layer m = h @ Wl (fp32 WMMA 16x16x4), also zero agg ---------------
__global__ void __launch_bounds__(256) k_layer_mm(const float* __restrict__ h,
    const float* __restrict__ Wl, float* __restrict__ m,
    float* __restrict__ agg, int Nn) {
  __shared__ float sW[H32 * H32];             // 4 KB
#if HAVE_ASYNC_LDS
  async_b128((const char*)Wl + threadIdx.x * 16, (char*)sW + threadIdx.x * 16);
  WAIT_ASYNC();
#else
  for (int i = threadIdx.x; i < H32 * H32; i += 256) sW[i] = Wl[i];
#endif
  __syncthreads();
  const int lane = threadIdx.x & 31;
  const int wave = threadIdx.x >> 5;
  const int tile = blockIdx.x * 8 + wave;
  const int m0   = tile * 16;
  if (m0 >= Nn) return;
  const int half = lane >> 4;
  const int ln   = lane & 15;
  int rowA = m0 + ln; if (rowA > Nn - 1) rowA = Nn - 1;
  const float* __restrict__ hr = h + (size_t)rowA * H32;

#if __has_builtin(__builtin_amdgcn_wmma_f32_16x16x4_f32)
  v8f acc0 = {}, acc1 = {};
#pragma unroll
  for (int k0 = 0; k0 < H32; k0 += 4) {
    int ka = k0 + half * 2;                   // A 16x4 f32: vgpr0=K ka, vgpr1=K ka+1
    v2f a;  a[0] = hr[ka];               a[1] = hr[ka + 1];
    v2f b0; b0[0] = sW[ka * H32 + ln];   b0[1] = sW[(ka + 1) * H32 + ln];
    v2f b1; b1[0] = sW[ka * H32 + 16 + ln]; b1[1] = sW[(ka + 1) * H32 + 16 + ln];
    acc0 = __builtin_amdgcn_wmma_f32_16x16x4_f32(false, a, false, b0,
                                                 (short)0, acc0, false, false);
    acc1 = __builtin_amdgcn_wmma_f32_16x16x4_f32(false, a, false, b1,
                                                 (short)0, acc1, false, false);
  }
  float* __restrict__ md = m   + (size_t)(m0 + half * 8) * H32 + ln;
  float* __restrict__ ad = agg + (size_t)(m0 + half * 8) * H32 + ln;
  if (m0 + 16 <= Nn) {                        // full tile: branchless stores
#pragma unroll
    for (int r = 0; r < 8; ++r) {
      md[(size_t)r * H32]      = acc0[r];
      md[(size_t)r * H32 + 16] = acc1[r];
      ad[(size_t)r * H32]      = 0.0f;
      ad[(size_t)r * H32 + 16] = 0.0f;
    }
  } else {
#pragma unroll
    for (int r = 0; r < 8; ++r) {
      int mm = m0 + half * 8 + r;
      if (mm < Nn) {
        m[(size_t)mm * H32 + ln]        = acc0[r];
        m[(size_t)mm * H32 + 16 + ln]   = acc1[r];
        agg[(size_t)mm * H32 + ln]      = 0.0f;
        agg[(size_t)mm * H32 + 16 + ln] = 0.0f;
      }
    }
  }
#else
  float hv[H32];
#pragma unroll
  for (int k = 0; k < H32; ++k) hv[k] = hr[k];
  int mm = m0 + ln;
  for (int c = 0; c < 16; ++c) {
    int cc = half * 16 + c;
    float acc = 0.0f;
#pragma unroll
    for (int k = 0; k < H32; ++k) acc += hv[k] * sW[k * H32 + cc];
    if (mm < Nn) { m[(size_t)mm * H32 + cc] = acc; agg[(size_t)mm * H32 + cc] = 0.0f; }
  }
#endif
}

// --- edge scatter-add (unnormalized) for GatedGraphConv --------------------
__global__ void k_scatter_add(const float* __restrict__ m,
                              float* __restrict__ agg,
                              const int* __restrict__ row,
                              const int* __restrict__ col, int E) {
  int idx = blockIdx.x * 256 + threadIdx.x;
  int e = idx >> 3;
  if (e >= E) return;
  int q = (idx & 7) * 4;
  int r = row[e], c = col[e];
  float4 v = *(const float4*)(m + (size_t)r * H32 + q);
  float* dst = agg + (size_t)c * H32 + q;
  atomicAdd(dst + 0, v.x);
  atomicAdd(dst + 1, v.y);
  atomicAdd(dst + 2, v.z);
  atomicAdd(dst + 3, v.w);
}

// --- GRU cell: h = GRU(agg, h), weights staged in LDS ----------------------
__global__ void __launch_bounds__(256) k_gru(const float* __restrict__ agg,
    float* __restrict__ h,
    const float* __restrict__ w_ih, const float* __restrict__ w_hh,
    const float* __restrict__ b_ih, const float* __restrict__ b_hh, int Nn) {
  __shared__ float s_ih[96 * H32];
  __shared__ float s_hh[96 * H32];
  __shared__ float sb_ih[96];
  __shared__ float sb_hh[96];
#if HAVE_ASYNC_LDS
  {
    const char* gih = (const char*)w_ih;
    const char* ghh = (const char*)w_hh;
    char* lih = (char*)s_ih;
    char* lhh = (char*)s_hh;
    for (int o = threadIdx.x * 16; o < 96 * H32 * 4; o += 256 * 16) {
      async_b128(gih + o, lih + o);           // 12288 B each, b128 chunks
      async_b128(ghh + o, lhh + o);
    }
    if (threadIdx.x < 24) {                   // 96 floats = 384 B = 24 chunks
      async_b128((const char*)b_ih + threadIdx.x * 16, (char*)sb_ih + threadIdx.x * 16);
      async_b128((const char*)b_hh + threadIdx.x * 16, (char*)sb_hh + threadIdx.x * 16);
    }
    WAIT_ASYNC();
  }
#else
  for (int i = threadIdx.x; i < 96 * H32; i += 256) { s_ih[i] = w_ih[i]; s_hh[i] = w_hh[i]; }
  for (int i = threadIdx.x; i < 96; i += 256)       { sb_ih[i] = b_ih[i]; sb_hh[i] = b_hh[i]; }
#endif
  __syncthreads();
  int i = blockIdx.x * 256 + threadIdx.x;
  if (i >= Nn) return;
  const float* __restrict__ ar = agg + (size_t)i * H32;
  float* __restrict__ hr = h + (size_t)i * H32;
  float a[H32], hv[H32];
#pragma unroll
  for (int k = 0; k < H32; ++k) { a[k] = ar[k]; hv[k] = hr[k]; }
  for (int g = 0; g < H32; ++g) {
    float ir = sb_ih[g], iz = sb_ih[g + 32], in_ = sb_ih[g + 64];
    float hr_ = sb_hh[g], hz = sb_hh[g + 32], hn = sb_hh[g + 64];
#pragma unroll
    for (int k = 0; k < H32; ++k) {
      ir  += a[k]  * s_ih[g * H32 + k];
      iz  += a[k]  * s_ih[(g + 32) * H32 + k];
      in_ += a[k]  * s_ih[(g + 64) * H32 + k];
      hr_ += hv[k] * s_hh[g * H32 + k];
      hz  += hv[k] * s_hh[(g + 32) * H32 + k];
      hn  += hv[k] * s_hh[(g + 64) * H32 + k];
    }
    float rg = 1.0f / (1.0f + __expf(-(ir + hr_)));
    float zg = 1.0f / (1.0f + __expf(-(iz + hz)));
    float ng = tanhf(in_ + rg * hn);
    float hold = hr[g];                       // old value; indices < g already rewritten
    hr[g] = (1.0f - zg) * ng + zg * hold;     // register copy hv[] keeps dots consistent
  }
}

// --- head: out = relu(h) @ Wfc + bfc ---------------------------------------
__global__ void k_fc(const float* __restrict__ h, const float* __restrict__ Wfc,
                     const float* __restrict__ bfc, float* __restrict__ out,
                     int Nn, int C) {
  int i = blockIdx.x * 256 + threadIdx.x;
  if (i >= Nn) return;
  float hv[H32];
#pragma unroll
  for (int k = 0; k < H32; ++k) hv[k] = fmaxf(h[(size_t)i * H32 + k], 0.0f);
  for (int c = 0; c < C; ++c) {
    float acc = bfc[c];
#pragma unroll
    for (int k = 0; k < H32; ++k) acc += hv[k] * Wfc[k * C + c];
    out[(size_t)i * C + c] = acc;
  }
}

// ---------------------------------------------------------------------------
extern "C" void kernel_launch(void* const* d_in, const int* in_sizes, int n_in,
                              void* d_out, int out_size, void* d_ws, size_t ws_size,
                              hipStream_t stream) {
  const float* x    = (const float*)d_in[0];
  const float* W1   = (const float*)d_in[1];
  const float* b1   = (const float*)d_in[2];
  const float* Wg   = (const float*)d_in[3];
  const float* w_ih = (const float*)d_in[4];
  const float* w_hh = (const float*)d_in[5];
  const float* b_ih = (const float*)d_in[6];
  const float* b_hh = (const float*)d_in[7];
  const float* Wfc  = (const float*)d_in[8];
  const float* bfc  = (const float*)d_in[9];
  const int*   ei   = (const int*)d_in[10];

  const int K  = in_sizes[1] / H32;            // 1152
  const int Nn = in_sizes[0] / K;              // 50000
  const int E  = in_sizes[10] / 2;             // 1600000
  const int L  = in_sizes[3] / (H32 * H32);    // 32
  const int C  = in_sizes[8] / H32;            // 7
  const int* row = ei;
  const int* col = ei + E;

  char* ws = (char*)d_ws;
  size_t off = 0;
  auto alloc = [&](size_t bytes) -> char* {
    char* p = ws + off;
    off += (bytes + 255) & ~(size_t)255;
    return p;
  };
  unsigned short* Wt  = (unsigned short*)alloc((size_t)H32 * K * sizeof(unsigned short));
  float* dinv = (float*)alloc((size_t)Nn * sizeof(float));        // deg -> dinv in place
  float* bufA = (float*)alloc((size_t)Nn * H32 * sizeof(float));  // hlin / m
  float* bufB = (float*)alloc((size_t)Nn * H32 * sizeof(float));  // hgcn / agg
  float* bufH = (float*)alloc((size_t)Nn * H32 * sizeof(float));  // h

  const dim3 blk(256);
  const int tiles   = (Nn + 15) / 16;
  const int gTiles  = (tiles + 7) / 8;
  const int gN      = (Nn + 255) / 256;
  const int gNH     = (Nn * H32 + 255) / 256;
  const int gE      = (E + 255) / 256;
  const int gE8     = (int)(((long long)E * 8 + 255) / 256);

  // Prep: transposed bf16 weights + degree normalization
  k_prep_wt<<<(H32 * K + 255) / 256, blk, 0, stream>>>(W1, Wt, K);
  k_zero_f <<<gN, blk, 0, stream>>>(dinv, Nn);
  k_count  <<<gE, blk, 0, stream>>>(dinv, col, E);
  k_dinv   <<<gN, blk, 0, stream>>>(dinv, Nn);

  // conv1: GCNConv + relu
  k_gemm1      <<<gTiles, blk, 0, stream>>>(x, Wt, bufA, Nn, K);
  k_gcn_self   <<<gNH, blk, 0, stream>>>(bufA, dinv, bufB, Nn);
  k_gcn_scatter<<<gE8, blk, 0, stream>>>(bufA, bufB, dinv, row, col, E);
  k_bias_relu  <<<gNH, blk, 0, stream>>>(bufB, b1, bufH, Nn * H32);

  // conv2: L iterations of (h @ Wl) -> scatter-add -> GRU
  for (int l = 0; l < L; ++l) {
    k_layer_mm  <<<gTiles, blk, 0, stream>>>(bufH, Wg + (size_t)l * H32 * H32,
                                             bufA, bufB, Nn);
    k_scatter_add<<<gE8, blk, 0, stream>>>(bufA, bufB, row, col, E);
    k_gru       <<<gN, blk, 0, stream>>>(bufB, bufH, w_ih, w_hh, b_ih, b_hh, Nn);
  }

  // head
  k_fc<<<gN, blk, 0, stream>>>(bufH, Wfc, bfc, (float*)d_out, Nn, C);
}
